// MultimodalCNNLSTMMOE_16260746182982
// MI455X (gfx1250) — matrix operand order repaired and checked by reading, not compiled, not run on hardware
//
#include <hip/hip_runtime.h>
#include <hip/hip_bf16.h>

// ---------------- problem sizes (from reference) ----------------
#define BATCH   4096
#define IN_DIM  1024
#define HID     512
#define OUT_DIM 512
#define NEXP    16
#define CTX     256
#define DEMO    16
// gate input = CTX + IN_DIM + DEMO = 1296

// ---------------- WMMA types ----------------
typedef __attribute__((ext_vector_type(16))) __bf16 v16bf;
typedef __attribute__((ext_vector_type(8)))  float  v8f;

union BFrag {
    v16bf        v;
    unsigned int u[8];
};

__device__ __forceinline__ unsigned short f2bf(float f) {
    unsigned int x = __float_as_uint(f);
    // round-to-nearest-even truncation to bf16
    unsigned int r = x + 0x7FFFu + ((x >> 16) & 1u);
    return (unsigned short)(r >> 16);
}

// =====================================================================
// Gate: logits -> softmax weights (ws), gate_usage (atomic), argmax
// =====================================================================
__global__ __launch_bounds__(256)
void gate_kernel(const float* __restrict__ x,
                 const float* __restrict__ u,
                 const float* __restrict__ d,
                 const float* __restrict__ gW,   // (1296, 16)
                 const float* __restrict__ gb,   // (16,)
                 float* __restrict__ weights,    // (B, 16) workspace
                 float* __restrict__ gate_usage, // (16,) in d_out (zeroed)
                 int*   __restrict__ routing)    // (B,) in d_out
{
    int b = blockIdx.x * blockDim.x + threadIdx.x;
    if (b >= BATCH) return;

    float logit[NEXP];
#pragma unroll
    for (int e = 0; e < NEXP; ++e) logit[e] = gb[e];

    // concat order: [u, x, d]
    for (int i = 0; i < CTX; ++i) {
        float v = u[b * CTX + i];
        const float* w = gW + (size_t)i * NEXP;
#pragma unroll
        for (int e = 0; e < NEXP; ++e) logit[e] += v * w[e];
    }
    for (int i = 0; i < IN_DIM; ++i) {
        float v = x[b * IN_DIM + i];
        const float* w = gW + (size_t)(CTX + i) * NEXP;
#pragma unroll
        for (int e = 0; e < NEXP; ++e) logit[e] += v * w[e];
    }
    for (int i = 0; i < DEMO; ++i) {
        float v = d[b * DEMO + i];
        const float* w = gW + (size_t)(CTX + IN_DIM + i) * NEXP;
#pragma unroll
        for (int e = 0; e < NEXP; ++e) logit[e] += v * w[e];
    }

    // softmax + argmax
    float m = logit[0];
    int   amax = 0;
#pragma unroll
    for (int e = 1; e < NEXP; ++e)
        if (logit[e] > m) { m = logit[e]; amax = e; }
    float s = 0.f;
    float ex[NEXP];
#pragma unroll
    for (int e = 0; e < NEXP; ++e) { ex[e] = __expf(logit[e] - m); s += ex[e]; }
    float inv = 1.f / s;
#pragma unroll
    for (int e = 0; e < NEXP; ++e) {
        float w = ex[e] * inv;
        weights[b * NEXP + e] = w;
        atomicAdd(&gate_usage[e], w * (1.f / (float)BATCH));
    }
    routing[b] = amax;
}

// =====================================================================
// x (B, IN_DIM) f32  ->  bf16 row-major (one-time per launch)
// =====================================================================
__global__ __launch_bounds__(256)
void cvt_x_kernel(const float* __restrict__ x, unsigned short* __restrict__ xb)
{
    size_t i = ((size_t)blockIdx.x * 256 + threadIdx.x) * 4;
    float4 v = *(const float4*)(x + i);
    union { unsigned short s[4]; uint2 u; } o;
    o.s[0] = f2bf(v.x); o.s[1] = f2bf(v.y);
    o.s[2] = f2bf(v.z); o.s[3] = f2bf(v.w);
    *(uint2*)(xb + i) = o.u;
}

// =====================================================================
// W (E, K, N) f32 -> Wt (E, N, K) bf16 (LDS-tiled transpose+convert)
// grid = (N/32, K/32, E), block = 256
// =====================================================================
__global__ __launch_bounds__(256)
void transpose_cvt_kernel(const float* __restrict__ src,
                          unsigned short* __restrict__ dst,
                          int K, int N)
{
    __shared__ float tile[32][33];
    const int e  = blockIdx.z;
    const int n0 = blockIdx.x * 32;
    const int k0 = blockIdx.y * 32;
    const int tx = threadIdx.x & 31;
    const int ty = threadIdx.x >> 5;      // 0..7
    const float* s = src + (size_t)e * K * N;
    unsigned short* d = dst + (size_t)e * N * K;

#pragma unroll
    for (int i = 0; i < 4; ++i) {
        int ky = ty + i * 8;
        tile[ky][tx] = s[(size_t)(k0 + ky) * N + n0 + tx];
    }
    __syncthreads();
#pragma unroll
    for (int i = 0; i < 4; ++i) {
        int ny = ty + i * 8;
        d[(size_t)(n0 + ny) * K + k0 + tx] = f2bf(tile[tx][ny]);
    }
}

// =====================================================================
// Fused per-expert MLP:  h = relu(x@W1[e]+b1) ; o = h@W2[e]+b2
// out[b,:] += weights[b,e] * o   (atomic accumulation across experts)
//
// grid  = (BATCH/32, NEXP), block = 256 (8 waves)
// wave layout: 2 (M) x 4 (N); each wave owns 16 rows x 128 cols
// (8 accumulator tiles) -> full N=512 in a single K pass.
//
// A/B fragments are direct global_load_b128 pairs from pre-converted
// bf16 buffers, software-pipelined with double buffers so each WMMA
// waits only on its own fragment (s_wait_loadcnt != 0 drain).
// LDS is used only to transpose h between the two GEMMs.
// =====================================================================
__global__ __launch_bounds__(256)
void moe_expert_kernel(const unsigned short* __restrict__ xb,  // (B,1024) bf16
                       const unsigned short* __restrict__ w1t, // (E,512,1024) bf16 [n][k]
                       const unsigned short* __restrict__ w2t, // (E,512,512)  bf16 [n][k]
                       const float* __restrict__ b1,           // (E, 512)
                       const float* __restrict__ b2,           // (E, 512)
                       const float* __restrict__ gw,           // (B, 16) softmax weights
                       float* __restrict__ out)                // (B, 512), zero-inited
{
    __shared__ unsigned short lds_h[32 * 520];   // h: 32 rows x 512, pitch 520 (16B-aligned rows)

    const int m0   = blockIdx.x * 32;
    const int e    = blockIdx.y;
    const int tid  = threadIdx.x;
    const int lane = tid & 31;
    const int wave = tid >> 5;
    const int wm   = wave & 1;        // row group of 16
    const int wn   = wave >> 1;       // 128-col group
    const int mg   = wm * 16;
    const int nl   = lane & 15;
    const int hi   = lane >> 4;       // lane half
    const int koff = hi * 8;          // A-frag K offset for upper lanes
    const int kbB  = hi * 8;          // B-frag K base for upper lanes

    const unsigned short* w1e = w1t + (size_t)e * HID * IN_DIM;
    const unsigned short* w2e = w2t + (size_t)e * OUT_DIM * HID;
    const float* b1e = b1 + (size_t)e * HID;
    const float* b2e = b2 + (size_t)e * OUT_DIM;

    v8f acc[8];
    BFrag a[2], b[2];

    // ---------------- GEMM1: h = relu(x @ W1e + b1e) ----------------
#pragma unroll
    for (int t = 0; t < 8; ++t)
#pragma unroll
        for (int i = 0; i < 8; ++i) acc[t][i] = 0.f;

    const unsigned short* arow = xb + (size_t)(m0 + mg + nl) * IN_DIM;

    auto loadA1 = [&](int k0, BFrag& dst) {
        *(uint4*)&dst.u[0] = *(const uint4*)(arow + k0 + koff);
        *(uint4*)&dst.u[4] = *(const uint4*)(arow + k0 + koff + 16);
    };
    auto loadB1 = [&](int k0, int t, BFrag& dst) {
        const unsigned short* bp =
            w1e + (size_t)(wn * 128 + t * 16 + nl) * IN_DIM + k0 + kbB;
        *(uint4*)&dst.u[0] = *(const uint4*)(bp);
        *(uint4*)&dst.u[4] = *(const uint4*)(bp + 16);
    };

    loadA1(0, a[0]);
    loadB1(0, 0, b[0]);

#pragma unroll 2
    for (int ks = 0; ks < IN_DIM / 32; ++ks) {
        const int k0 = ks * 32;
        const int kn = (ks + 1 < IN_DIM / 32) ? k0 + 32 : k0;  // clamp at tail
        __builtin_prefetch(arow + k0 + 256, 0, 1);             // speculative
#pragma unroll
        for (int t = 0; t < 8; ++t) {
            if (t < 7) {
                loadB1(k0, t + 1, b[(t + 1) & 1]);
            } else {
                loadB1(kn, 0, b[0]);             // next k-step, tile 0
                loadA1(kn, a[(ks + 1) & 1]);     // next k-step A frag
            }
            acc[t] = __builtin_amdgcn_wmma_f32_16x16x32_bf16(
                false, a[ks & 1].v, false, b[t & 1].v, (short)0, acc[t],
                false, false);
        }
    }

    // epilogue: +b1, ReLU, store bf16 h tile to LDS (C-layout -> row-major)
#pragma unroll
    for (int t = 0; t < 8; ++t) {
        const int n = wn * 128 + t * 16 + nl;
        const float bias = b1e[n];
#pragma unroll
        for (int r = 0; r < 8; ++r) {
            int ml = mg + r + hi * 8;
            float v = acc[t][r] + bias;
            v = v > 0.f ? v : 0.f;
            lds_h[ml * 520 + n] = f2bf(v);
        }
    }
    __syncthreads();

    // ---------------- GEMM2: o = h @ W2e ; out += w * (o + b2) ------
#pragma unroll
    for (int t = 0; t < 8; ++t)
#pragma unroll
        for (int i = 0; i < 8; ++i) acc[t][i] = 0.f;

    const unsigned short* hrow = &lds_h[(mg + nl) * 520];

    auto loadA2 = [&](int k0, BFrag& dst) {
        *(uint4*)&dst.u[0] = *(const uint4*)(hrow + k0 + koff);
        *(uint4*)&dst.u[4] = *(const uint4*)(hrow + k0 + koff + 16);
    };
    auto loadB2 = [&](int k0, int t, BFrag& dst) {
        const unsigned short* bp =
            w2e + (size_t)(wn * 128 + t * 16 + nl) * HID + k0 + kbB;
        *(uint4*)&dst.u[0] = *(const uint4*)(bp);
        *(uint4*)&dst.u[4] = *(const uint4*)(bp + 16);
    };

    loadA2(0, a[0]);
    loadB2(0, 0, b[0]);

#pragma unroll 2
    for (int ks = 0; ks < HID / 32; ++ks) {
        const int k0 = ks * 32;
        const int kn = (ks + 1 < HID / 32) ? k0 + 32 : k0;
#pragma unroll
        for (int t = 0; t < 8; ++t) {
            if (t < 7) {
                loadB2(k0, t + 1, b[(t + 1) & 1]);
            } else {
                loadB2(kn, 0, b[0]);
                loadA2(kn, a[(ks + 1) & 1]);
            }
            acc[t] = __builtin_amdgcn_wmma_f32_16x16x32_bf16(
                false, a[ks & 1].v, false, b[t & 1].v, (short)0, acc[t],
                false, false);
        }
    }

    // epilogue: gate-weighted accumulation into global output
    float wr[8];
#pragma unroll
    for (int r = 0; r < 8; ++r)
        wr[r] = gw[(size_t)(m0 + mg + hi * 8 + r) * NEXP + e];

#pragma unroll
    for (int t = 0; t < 8; ++t) {
        const int n = wn * 128 + t * 16 + nl;
        const float bias = b2e[n];
#pragma unroll
        for (int r = 0; r < 8; ++r) {
            int m = m0 + mg + hi * 8 + r;
            float v = (acc[t][r] + bias) * wr[r];
            atomicAdd(&out[(size_t)m * OUT_DIM + n], v);
        }
    }
}

// =====================================================================
extern "C" void kernel_launch(void* const* d_in, const int* in_sizes, int n_in,
                              void* d_out, int out_size, void* d_ws, size_t ws_size,
                              hipStream_t stream) {
    const float* x  = (const float*)d_in[0];
    const float* u  = (const float*)d_in[1];
    const float* dd = (const float*)d_in[2];
    const float* gW = (const float*)d_in[3];
    const float* gb = (const float*)d_in[4];
    const float* W1 = (const float*)d_in[5];
    const float* b1 = (const float*)d_in[6];
    const float* W2 = (const float*)d_in[7];
    const float* b2 = (const float*)d_in[8];

    float* out        = (float*)d_out;                      // (B, OUT)
    float* gate_usage = out + (size_t)BATCH * OUT_DIM;      // (16,)
    int*   routing    = (int*)(gate_usage + NEXP);          // (B,)

    // workspace layout (bytes):
    //   [0)                weights (B*16 f32)          = 256 KB
    //   [256K)             xb  (B*IN_DIM bf16)         = 8 MB
    //   [256K+8M)          w1t (E*HID*IN_DIM bf16)     = 16 MB
    //   [256K+24M)         w2t (E*OUT*HID bf16)        = 8 MB
    char* ws = (char*)d_ws;
    float*          weights = (float*)ws;
    unsigned short* xb      = (unsigned short*)(ws + 262144);
    unsigned short* w1t     = (unsigned short*)(ws + 262144 + 8388608);
    unsigned short* w2t     = (unsigned short*)(ws + 262144 + 8388608 + 16777216);

    // zero the atomic-accumulated regions (out + gate_usage)
    hipMemsetAsync(d_out, 0,
                   ((size_t)BATCH * OUT_DIM + NEXP) * sizeof(float), stream);

    gate_kernel<<<BATCH / 256, 256, 0, stream>>>(
        x, u, dd, gW, gb, weights, gate_usage, routing);

    // one-time (per launch) precision conversion / transposition
    cvt_x_kernel<<<(BATCH * IN_DIM / 4) / 256, 256, 0, stream>>>(x, xb);
    transpose_cvt_kernel<<<dim3(HID / 32, IN_DIM / 32, NEXP), 256, 0, stream>>>(
        W1, w1t, IN_DIM, HID);
    transpose_cvt_kernel<<<dim3(OUT_DIM / 32, HID / 32, NEXP), 256, 0, stream>>>(
        W2, w2t, HID, OUT_DIM);

    dim3 grid(BATCH / 32, NEXP);
    moe_expert_kernel<<<grid, 256, 0, stream>>>(
        xb, w1t, w2t, b1, b2, weights, out);
}
